// EdgeNetwork_66692252172957
// MI455X (gfx1250) — matrix operand order repaired and checked by reading, not compiled, not run on hardware
//
#include <hip/hip_runtime.h>

typedef __attribute__((ext_vector_type(16))) __bf16 v16bf;
typedef __attribute__((ext_vector_type(8)))  float v8f;
typedef __attribute__((ext_vector_type(4)))  int   v4i;

#define IN_DIM   256
#define HIDDEN   512
#define OUT_DIM  256
#define TILE_M   64
#define EF_STRIDE 520                        // bf16 elems per LDS row (512 + 8 pad)
#define EF_BYTES  (TILE_M * EF_STRIDE * 2)   // 66560
#define H_OFF     EF_BYTES
#define BST_OFF   (2 * EF_BYTES)             // 133120
#define BST_STRIDE 1536                      // bytes per staged fragment (32 lanes * 48B)
#define BST_BUF   (32 * BST_STRIDE)          // 49152 per double-buffer slot
#define LDS_TOTAL (BST_OFF + 2 * BST_BUF)    // 231424
#define W1_DWORDS (HIDDEN * HIDDEN / 2)      // 131072 packed bf16 pairs
#define W2_DWORDS (HIDDEN * OUT_DIM / 2)     // 65536

#if defined(__HIP_DEVICE_COMPILE__) && \
    __has_builtin(__builtin_amdgcn_global_load_async_to_lds_b128) && \
    __has_builtin(__builtin_amdgcn_s_wait_asynccnt)
#define USE_ASYNC 1
#else
#define USE_ASYNC 0
#endif

__device__ __forceinline__ unsigned short f2bf(float f) {
    unsigned int u = __float_as_uint(f);
    u += 0x7FFFu + ((u >> 16) & 1u);         // round-to-nearest-even
    return (unsigned short)(u >> 16);
}
__device__ __forceinline__ unsigned int pack2(float a, float b) {
    return (unsigned int)f2bf(a) | ((unsigned int)f2bf(b) << 16);
}

// 16-byte async copy global->LDS (ASYNCcnt path), sync fallback otherwise.
__device__ __forceinline__ void async_cp16(const unsigned int* g, unsigned int* l) {
#if USE_ASYNC
    __builtin_amdgcn_global_load_async_to_lds_b128(
        (__attribute__((address_space(1))) v4i*)g,
        (__attribute__((address_space(3))) v4i*)l, 0, 0);
#else
    *(uint4*)l = *(const uint4*)g;
#endif
}
__device__ __forceinline__ void async_wait0() {
#if USE_ASYNC
    __builtin_amdgcn_s_wait_asynccnt(0);
#endif
}

union Frag16 { v16bf v; uint4 q[2]; };

// ---------------------------------------------------------------------------
// Prep: reorder W1 [512x512] f32 and W2 [512x256] f32 into bf16 WMMA
// B-fragment layout: frag f = ktile*NT + ntile; dword (lane*8 + i) =
// pack( W[kb][col], W[kb+1][col] ), col = ntile*16+lane%16,
// kb = ktile*32 + (lane<16?0:16) + 2*i
// ---------------------------------------------------------------------------
__global__ void EdgeNet_prep_weights(const float* __restrict__ W1,
                                     const float* __restrict__ W2,
                                     unsigned int* __restrict__ w1f,
                                     unsigned int* __restrict__ w2f) {
    int d = blockIdx.x * 256 + threadIdx.x;
    if (d < W1_DWORDS) {
        int f = d >> 8, rem = d & 255;
        int lane = rem >> 3, i = rem & 7;
        int ktile = f >> 5, nt = f & 31;
        int col = nt * 16 + (lane & 15);
        int kb = ktile * 32 + ((lane < 16) ? 0 : 16) + 2 * i;
        w1f[d] = pack2(W1[kb * HIDDEN + col], W1[(kb + 1) * HIDDEN + col]);
    } else if (d < W1_DWORDS + W2_DWORDS) {
        int d2 = d - W1_DWORDS;
        int f = d2 >> 8, rem = d2 & 255;
        int lane = rem >> 3, i = rem & 7;
        int ktile = f >> 4, nt = f & 15;
        int col = nt * 16 + (lane & 15);
        int kb = ktile * 32 + ((lane < 16) ? 0 : 16) + 2 * i;
        w2f[d2] = pack2(W2[kb * OUT_DIM + col], W2[(kb + 1) * OUT_DIM + col]);
    }
}

// ---------------------------------------------------------------------------
// Main: 64 edges per block, 8 waves (wave32), async double-buffered B stage.
// ---------------------------------------------------------------------------
__global__ __launch_bounds__(256)
void EdgeNet_main(const float* __restrict__ x,
                  const long long* __restrict__ ei,
                  const float* __restrict__ b1,
                  const float* __restrict__ b2,
                  const unsigned int* __restrict__ w1f,
                  const unsigned int* __restrict__ w2f,
                  float* __restrict__ out, int E) {
    extern __shared__ char smem[];   // [ef 66560 | h 66560 | Bstage 2x49152]
    const int tid = threadIdx.x;

    // ---- Phase 0: gather + f32->bf16 convert into LDS ef[64][512] ----------
    {
        int row = tid >> 2;
        int tg  = tid & 3;
        long long e  = (long long)blockIdx.x * TILE_M + row;
        long long ec = (e < E) ? e : 0;
        long long ns = ei[ec];
        long long nt = ei[(long long)E + ec];
        long long node = (tg < 2) ? ns : nt;
        int colInNode = (tg & 1) * 128;
        const float4* src = (const float4*)(x + node * IN_DIM + colInNode);
        int ldsCol = ((tg < 2) ? 0 : IN_DIM) + colInNode;
        uint2* dst = (uint2*)(smem + row * (EF_STRIDE * 2) + ldsCol * 2);
        #pragma unroll 8
        for (int i = 0; i < 32; ++i) {
            float4 f = src[i];
            uint2 p;
            p.x = pack2(f.x, f.y);
            p.y = pack2(f.z, f.w);
            dst[i] = p;
        }
    }
    __syncthreads();

    const int wave  = tid >> 5;
    const int lane  = tid & 31;
    const int mslab = wave & 3;               // 16-row slab
    const int arow  = mslab * 16 + (lane & 15);
    const int abase = (lane < 16) ? 0 : 16;   // A-frag K sub-offset (elems)
    const int mhi   = (lane < 16) ? 0 : 8;    // C/D row offset for this lane
    const char* efp = smem;
    unsigned short* hls = (unsigned short*)(smem + H_OFF);
    const v8f vzero = {0.f, 0.f, 0.f, 0.f, 0.f, 0.f, 0.f, 0.f};

    // Cooperative B staging: wave w stages frags [w*4, w*4+4) (GEMM1, 32 frags)
    // or [w*2, w*2+2) (GEMM2, 16 frags). 48B lane stride -> bank-conflict-free.
    auto stage_w1 = [&](int ks, int buf) {
        char* base = smem + BST_OFF + buf * BST_BUF;
        #pragma unroll
        for (int j = 0; j < 4; ++j) {
            int nt = wave * 4 + j;
            const unsigned int* g = w1f + (ks * 32 + nt) * 256 + lane * 8;
            unsigned int* l = (unsigned int*)(base + nt * BST_STRIDE + lane * 48);
            async_cp16(g, l);
            async_cp16(g + 4, l + 4);
        }
    };
    auto stage_w2 = [&](int ks, int buf) {
        char* base = smem + BST_OFF + buf * BST_BUF;
        #pragma unroll
        for (int j = 0; j < 2; ++j) {
            int nt = wave * 2 + j;
            const unsigned int* g = w2f + (ks * 16 + nt) * 256 + lane * 8;
            unsigned int* l = (unsigned int*)(base + nt * BST_STRIDE + lane * 48);
            async_cp16(g, l);
            async_cp16(g + 4, l + 4);
        }
    };

    // ---- GEMM1: h[64x512] = relu(ef @ W1 + b1); wave = 16 rows x 256 cols --
    const int nhalf = wave >> 2;
    {
        v8f acc[16];
        #pragma unroll
        for (int t = 0; t < 16; ++t) acc[t] = vzero;

        stage_w1(0, 0);
        for (int ks = 0; ks < HIDDEN / 32; ++ks) {
            async_wait0();
            __syncthreads();                     // stage[ks] visible to all
            if (ks < HIDDEN / 32 - 1) stage_w1(ks + 1, (ks + 1) & 1);

            const int kk = ks * 32;
            Frag16 a;
            const uint4* ap =
                (const uint4*)(efp + (arow * EF_STRIDE + kk + abase) * 2);
            a.q[0] = ap[0];                      // K = base .. base+7
            a.q[1] = ap[1];                      // K = base+8 .. base+15
            const char* bb = smem + BST_OFF + (ks & 1) * BST_BUF;
            #pragma unroll
            for (int t = 0; t < 16; ++t) {
                const uint4* bp =
                    (const uint4*)(bb + (nhalf * 16 + t) * BST_STRIDE + lane * 48);
                Frag16 b;
                b.q[0] = bp[0];
                b.q[1] = bp[1];
                acc[t] = __builtin_amdgcn_wmma_f32_16x16x32_bf16(
                    false, a.v, false, b.v, (short)0, acc[t], false, false);
            }
        }
        // bias + relu + bf16 store of h tiles
        #pragma unroll
        for (int t = 0; t < 16; ++t) {
            int ncol = nhalf * 256 + t * 16 + (lane & 15);
            float bias = b1[ncol];
            #pragma unroll
            for (int r = 0; r < 8; ++r) {
                float v = acc[t][r] + bias;
                v = v > 0.f ? v : 0.f;
                int hrow = mslab * 16 + mhi + r;
                hls[hrow * EF_STRIDE + ncol] = f2bf(v);
            }
        }
    }

    // ---- GEMM2: out[64x256] = h @ W2 + b2; wave = 16 rows x 128 cols -------
    const int nq = wave >> 2;
    {
        v8f acc2[8];
        #pragma unroll
        for (int t = 0; t < 8; ++t) acc2[t] = vzero;

        stage_w2(0, 0);                          // safe: all waves past ks=15 barrier
        const char* hp = (const char*)hls;
        for (int ks = 0; ks < HIDDEN / 32; ++ks) {
            async_wait0();
            __syncthreads();                     // stage[ks] + (ks==0) h visible
            if (ks < HIDDEN / 32 - 1) stage_w2(ks + 1, (ks + 1) & 1);

            const int kk = ks * 32;
            Frag16 a;
            const uint4* ap =
                (const uint4*)(hp + (arow * EF_STRIDE + kk + abase) * 2);
            a.q[0] = ap[0];
            a.q[1] = ap[1];
            const char* bb = smem + BST_OFF + (ks & 1) * BST_BUF;
            #pragma unroll
            for (int t = 0; t < 8; ++t) {
                const uint4* bp =
                    (const uint4*)(bb + (nq * 8 + t) * BST_STRIDE + lane * 48);
                Frag16 b;
                b.q[0] = bp[0];
                b.q[1] = bp[1];
                acc2[t] = __builtin_amdgcn_wmma_f32_16x16x32_bf16(
                    false, a.v, false, b.v, (short)0, acc2[t], false, false);
            }
        }
        #pragma unroll
        for (int t = 0; t < 8; ++t) {
            int ncol = nq * 128 + t * 16 + (lane & 15);
            float bias = b2[ncol];
            #pragma unroll
            for (int r = 0; r < 8; ++r) {
                int rowl = mslab * 16 + mhi + r;
                long long e = (long long)blockIdx.x * TILE_M + rowl;
                if (e < E) out[e * OUT_DIM + ncol] = acc2[t][r] + bias;
            }
        }
    }
}

extern "C" void kernel_launch(void* const* d_in, const int* in_sizes, int n_in,
                              void* d_out, int out_size, void* d_ws, size_t ws_size,
                              hipStream_t stream) {
    const float*     x  = (const float*)d_in[0];
    const long long* ei = (const long long*)d_in[1];
    const float*     W1 = (const float*)d_in[2];
    const float*     b1 = (const float*)d_in[3];
    const float*     W2 = (const float*)d_in[4];
    const float*     b2 = (const float*)d_in[5];
    float* out = (float*)d_out;
    const int E = in_sizes[1] / 2;

    unsigned int* w1f = (unsigned int*)d_ws;
    unsigned int* w2f = w1f + W1_DWORDS;

    int prepThreads = W1_DWORDS + W2_DWORDS;
    EdgeNet_prep_weights<<<(prepThreads + 255) / 256, 256, 0, stream>>>(W1, W2, w1f, w2f);

    int nBlocks = (E + TILE_M - 1) / TILE_M;   // 5000 for E=320000
    EdgeNet_main<<<nBlocks, 256, LDS_TOTAL, stream>>>(x, ei, b1, b2, w1f, w2f, out, E);
}
